// DnnDependencyParser_53197464928367
// MI455X (gfx1250) — compile-verified, matrix operand order checked by reading, not run on hardware
//
#include <hip/hip_runtime.h>

// ---------------------------------------------------------------------------
// Types for CDNA5 WMMA (wave32, v_wmma_f32_16x16x32_bf16)
// ---------------------------------------------------------------------------
typedef __bf16 bf16;
typedef __attribute__((ext_vector_type(16))) __bf16 v16bf;
typedef __attribute__((ext_vector_type(8)))  __bf16 v8bf;
typedef __attribute__((ext_vector_type(8)))  float   v8f;

static constexpr int TOK  = 768;    // sequence length N
static constexpr int HID  = 512;    // LSTM hidden per direction
static constexpr int G4   = 2048;   // 4*HID gate width
static constexpr int KP1  = 128;    // layer-1 input dim padded (125 -> 128)
static constexpr int KP2  = 1024;   // layer-2 input dim (2*HID)
static constexpr int DMP  = 128;    // MLP dim padded (100 -> 128)
static constexpr int DMV  = 100;    // valid MLP dim

// ---------------------------------------------------------------------------
// fp32 -> bf16 convert with zero padding: dst[Rp,Cp] <- src[R,C]
// ---------------------------------------------------------------------------
__global__ void cvt_pad_bf16(const float* __restrict__ src, bf16* __restrict__ dst,
                             int R, int C, int Rp, int Cp) {
    int i = blockIdx.x * blockDim.x + threadIdx.x;
    if (i >= Rp * Cp) return;
    int r = i / Cp, c = i % Cp;
    float v = (r < R && c < C) ? src[r * C + c] : 0.f;
    dst[i] = (bf16)v;
}

// fp32 vector pad: dst[np] <- src[n], zeros beyond
__global__ void pad_f32(const float* __restrict__ src, float* __restrict__ dst,
                        int n, int np) {
    int i = blockIdx.x * blockDim.x + threadIdx.x;
    if (i < np) dst[i] = (i < n) ? src[i] : 0.f;
}

// dst = a + b (fp32), length n   (combined LSTM bias bih + bhh)
__global__ void vec_add(const float* __restrict__ a, const float* __restrict__ b,
                        float* __restrict__ dst, int n) {
    int i = blockIdx.x * blockDim.x + threadIdx.x;
    if (i < n) dst[i] = a[i] + b[i];
}

// ---------------------------------------------------------------------------
// Embedding gather + concat -> x0 bf16 [TOK, 128]  (cols 125..127 zero)
// ---------------------------------------------------------------------------
__global__ void embed_concat(const int* __restrict__ widx, const int* __restrict__ pidx,
                             const float* __restrict__ wemb, const float* __restrict__ pemb,
                             bf16* __restrict__ x0) {
    int n = blockIdx.x;         // token
    int c = threadIdx.x;        // 0..127
    float v = 0.f;
    if (c < 100)       v = wemb[widx[n] * 100 + c];
    else if (c < 125)  v = pemb[pidx[n] * 25 + (c - 100)];
    x0[n * KP1 + c] = (bf16)v;
}

// concat fwd/bwd hidden states, cast to bf16: x[TOK, 1024]
__global__ void concat_bf16(const float* __restrict__ hf, const float* __restrict__ hb,
                            bf16* __restrict__ x) {
    int i = blockIdx.x * blockDim.x + threadIdx.x;   // over TOK*HID
    if (i >= TOK * HID) return;
    int t = i >> 9, j = i & (HID - 1);
    x[t * (2 * HID) + j]       = (bf16)hf[i];
    x[t * (2 * HID) + HID + j] = (bf16)hb[i];
}

// ---------------------------------------------------------------------------
// WMMA bf16 GEMM:  C[M,N] (fp32) = A[M,K] * B[N,K]^T (+ bias[n])
// A, B row-major bf16, K contiguous; K % 32 == 0, M % 64 == 0, N % 64 == 0.
// Block = 128 threads (4 waves). Each wave computes a 32x32 C tile (2x2 WMMA
// tiles, 4 accumulators) -> 64x64 per block. 4 fragment loads feed 4 WMMAs
// per K-step, halving redundant L2 fragment traffic vs 1 tile/wave.
//
// Per-lane fragment gather for both A and B (ISA 7.12.2, 16-bit 16x32 layout):
//   row  = lane & 15
//   kblk = (lane & 16) ? 8 : 0   -> halves [kblk..kblk+7] and [kblk+16..kblk+23]
// ---------------------------------------------------------------------------
__device__ __forceinline__ v16bf load_frag16x32(const bf16* __restrict__ base,
                                                int ld, int lane) {
    int r  = lane & 15;
    int kb = (lane & 16) ? 8 : 0;
    const bf16* p = base + r * ld + kb;
    v8bf lo = *(const v8bf*)(p);        // 16B
    v8bf hi = *(const v8bf*)(p + 16);   // 16B
    v16bf out;
#pragma unroll
    for (int i = 0; i < 8; ++i) { out[i] = lo[i]; out[i + 8] = hi[i]; }
    return out;
}

__device__ __forceinline__ v8f wmma_bf16(v16bf a, v16bf b, v8f c) {
    return __builtin_amdgcn_wmma_f32_16x16x32_bf16(
        /*neg_a=*/false, a, /*neg_b=*/false, b,
        /*c_mod=*/(short)0, c, /*reuse_a=*/false, /*reuse_b=*/false);
}

__global__ void gemm_bf16_wmma(const bf16* __restrict__ A, const bf16* __restrict__ B,
                               const float* __restrict__ bias, float* __restrict__ C,
                               int M, int N, int K, int ldc) {
    int wave = threadIdx.x >> 5;               // 0..3
    int lane = threadIdx.x & 31;
    int m0 = blockIdx.y * 64 + (wave >> 1) * 32;
    int n0 = blockIdx.x * 64 + (wave & 1) * 32;
    if (m0 >= M || n0 >= N) return;

    const bf16* A0 = A + (size_t)m0 * K;
    const bf16* A1 = A + (size_t)(m0 + 16) * K;
    const bf16* B0 = B + (size_t)n0 * K;
    const bf16* B1 = B + (size_t)(n0 + 16) * K;

    v8f acc00 = {}, acc01 = {}, acc10 = {}, acc11 = {};
    for (int k = 0; k < K; k += 32) {
        v16bf a0 = load_frag16x32(A0 + k, K, lane);
        v16bf a1 = load_frag16x32(A1 + k, K, lane);
        v16bf b0 = load_frag16x32(B0 + k, K, lane);
        v16bf b1 = load_frag16x32(B1 + k, K, lane);
        acc00 = wmma_bf16(a0, b0, acc00);
        acc01 = wmma_bf16(a0, b1, acc01);
        acc10 = wmma_bf16(a1, b0, acc10);
        acc11 = wmma_bf16(a1, b1, acc11);
    }

    int n  = lane & 15;
    int mo = (lane & 16) ? 8 : 0;
    float bv0 = bias ? bias[n0 + n]      : 0.f;
    float bv1 = bias ? bias[n0 + 16 + n] : 0.f;
#pragma unroll
    for (int v = 0; v < 8; ++v) {
        C[(size_t)(m0 + mo + v)      * ldc + n0 + n]      = acc00[v] + bv0;
        C[(size_t)(m0 + mo + v)      * ldc + n0 + 16 + n] = acc01[v] + bv1;
        C[(size_t)(m0 + 16 + mo + v) * ldc + n0 + n]      = acc10[v] + bv0;
        C[(size_t)(m0 + 16 + mo + v) * ldc + n0 + 16 + n] = acc11[v] + bv1;
    }
}

// ---------------------------------------------------------------------------
// Persistent LSTM scan: one 1024-thread block per direction.
//  gx: [TOK, 2048] fp32 (x@Wih^T + bih + bhh, gate order i,f,g,o)
//  W : [2048, 512] bf16 row-major (Whh)
//  hs: [TOK, 512] fp32 output hidden states
// Threads t<512: rows j (i-gate) and 512+j (f-gate); t>=512: 1024+j (g), 1536+j (o).
// h lives in LDS (read as float4 broadcasts); Whh streams from L2 (2 MB bf16).
// ---------------------------------------------------------------------------
__global__ __launch_bounds__(1024)
void lstm_scan(const float* __restrict__ gx_f, const float* __restrict__ gx_b,
               const bf16* __restrict__ W_f, const bf16* __restrict__ W_b,
               float* __restrict__ hs_f, float* __restrict__ hs_b) {
    const bool rev = (blockIdx.x == 1);
    const float* gx = rev ? gx_b : gx_f;
    const bf16*  W  = rev ? W_b  : W_f;
    float*       hs = rev ? hs_b : hs_f;

    __shared__ __align__(16) float h_sh[HID];
    __shared__ float g_sh[HID];
    __shared__ float o_sh[HID];

    const int t      = threadIdx.x;       // 0..1023
    const int j      = t & (HID - 1);
    const bool upper = (t >= HID);
    const int row0   = upper ? (2 * HID + j) : j;
    const int row1   = row0 + HID;
    const bf16* w0 = W + (size_t)row0 * HID;
    const bf16* w1 = W + (size_t)row1 * HID;

    float c = 0.f;
    if (t < HID) h_sh[t] = 0.f;
    __syncthreads();

    for (int s = 0; s < TOK; ++s) {
        const int tt = rev ? (TOK - 1 - s) : s;
        const float* g = gx + (size_t)tt * G4;
        float d0 = g[row0], d1 = g[row1];
        for (int k = 0; k < HID; k += 8) {
            v8bf a0 = *(const v8bf*)(w0 + k);          // 16B global
            v8bf a1 = *(const v8bf*)(w1 + k);          // 16B global
            float hbuf[8];
            *(float4*)&hbuf[0] = *(const float4*)(h_sh + k);      // ds b128
            *(float4*)&hbuf[4] = *(const float4*)(h_sh + k + 4);  // ds b128
#pragma unroll
            for (int u = 0; u < 8; ++u) {
                d0 += hbuf[u] * (float)a0[u];
                d1 += hbuf[u] * (float)a1[u];
            }
        }
        __syncthreads();                       // all h_sh reads done
        if (upper) { g_sh[j] = d0; o_sh[j] = d1; }
        __syncthreads();
        if (!upper) {
            float iv = 1.f / (1.f + __expf(-d0));
            float fv = 1.f / (1.f + __expf(-d1));
            float gv = tanhf(g_sh[j]);
            float ov = 1.f / (1.f + __expf(-o_sh[j]));
            c = fv * c + iv * gv;
            float h = ov * tanhf(c);
            h_sh[j] = h;
            hs[(size_t)tt * HID + j] = h;
        }
        __syncthreads();                       // new h_sh visible
    }
}

// ---------------------------------------------------------------------------
// scores[h,m] = b2 + sum_d w2[d] * relu(Hh[h,d] + Hm[m,d]),  d < 100
// Hh/Hm stored fp32 with ld = 128. Block (16,16), grid (48,48).
// ---------------------------------------------------------------------------
__global__ void biaffine_scores(const float* __restrict__ Hh, const float* __restrict__ Hm,
                                const float* __restrict__ w2, const float* __restrict__ b2,
                                float* __restrict__ out) {
    __shared__ float sh[16][104];
    __shared__ float sm[16][104];
    __shared__ float w2s[DMV];
    const int h0 = blockIdx.y * 16, m0 = blockIdx.x * 16;
    const int tx = threadIdx.x, ty = threadIdx.y;
    const int tid = ty * 16 + tx;

    for (int i = tid; i < 16 * DMV; i += 256) {
        int r = i / DMV, c = i % DMV;
        sh[r][c] = Hh[(size_t)(h0 + r) * DMP + c];
        sm[r][c] = Hm[(size_t)(m0 + r) * DMP + c];
    }
    if (tid < DMV) w2s[tid] = w2[tid];
    __syncthreads();

    float acc = b2[0];
    for (int d = 0; d < DMV; ++d) {
        float v = sh[ty][d] + sm[tx][d];
        acc += w2s[d] * fmaxf(v, 0.f);
    }
    out[(size_t)(h0 + ty) * TOK + (m0 + tx)] = acc;
}

// ---------------------------------------------------------------------------
// Host launcher
// ---------------------------------------------------------------------------
static inline int div_up(int a, int b) { return (a + b - 1) / b; }

extern "C" void kernel_launch(void* const* d_in, const int* in_sizes, int n_in,
                              void* d_out, int out_size, void* d_ws, size_t ws_size,
                              hipStream_t stream) {
    (void)in_sizes; (void)n_in; (void)out_size; (void)ws_size;
    // Input order (setup_inputs dict order, leaves in insertion order):
    //  0 word_idx[768] i32, 1 pos_idx[768] i32, 2 head[768] i32 (unused),
    //  3 word_emb[50000,100], 4 pos_emb[50,25],
    //  5.. lstm L0 fwd: Wih[2048,125], Whh[2048,512], bih[2048], bhh[2048]
    //  9.. lstm L0 bwd: same
    // 13.. lstm L1 fwd: Wih[2048,1024], Whh, bih, bhh
    // 17.. lstm L1 bwd: same
    // 21 Wh[100,1024], 22 bh[100], 23 Wm[100,1024], 24 W2[1,100], 25 b2[1]
    const int*   widx  = (const int*)d_in[0];
    const int*   pidx  = (const int*)d_in[1];
    const float* wemb  = (const float*)d_in[3];
    const float* pemb  = (const float*)d_in[4];
    const float* Wih1f = (const float*)d_in[5],  *Whh1f = (const float*)d_in[6];
    const float* bih1f = (const float*)d_in[7],  *bhh1f = (const float*)d_in[8];
    const float* Wih1b = (const float*)d_in[9],  *Whh1b = (const float*)d_in[10];
    const float* bih1b = (const float*)d_in[11], *bhh1b = (const float*)d_in[12];
    const float* Wih2f = (const float*)d_in[13], *Whh2f = (const float*)d_in[14];
    const float* bih2f = (const float*)d_in[15], *bhh2f = (const float*)d_in[16];
    const float* Wih2b = (const float*)d_in[17], *Whh2b = (const float*)d_in[18];
    const float* bih2b = (const float*)d_in[19], *bhh2b = (const float*)d_in[20];
    const float* Wh    = (const float*)d_in[21], *bh    = (const float*)d_in[22];
    const float* Wm    = (const float*)d_in[23];
    const float* W2    = (const float*)d_in[24], *b2    = (const float*)d_in[25];
    float* scores = (float*)d_out;

    // ---- workspace layout (deterministic bump allocator, 256B aligned) ----
    char* wsb = (char*)d_ws; size_t off = 0;
    auto alloc = [&](size_t bytes) -> char* {
        char* p = wsb + off; off = (off + bytes + 255) & ~(size_t)255; return p;
    };
    bf16* x0      = (bf16*)alloc((size_t)TOK * KP1 * 2);
    bf16* x1      = (bf16*)alloc((size_t)TOK * KP2 * 2);
    bf16* x2      = (bf16*)alloc((size_t)TOK * KP2 * 2);
    bf16* cWih1f  = (bf16*)alloc((size_t)G4 * KP1 * 2);
    bf16* cWih1b  = (bf16*)alloc((size_t)G4 * KP1 * 2);
    bf16* cWih2f  = (bf16*)alloc((size_t)G4 * KP2 * 2);
    bf16* cWih2b  = (bf16*)alloc((size_t)G4 * KP2 * 2);
    bf16* cWhh1f  = (bf16*)alloc((size_t)G4 * HID * 2);
    bf16* cWhh1b  = (bf16*)alloc((size_t)G4 * HID * 2);
    bf16* cWhh2f  = (bf16*)alloc((size_t)G4 * HID * 2);
    bf16* cWhh2b  = (bf16*)alloc((size_t)G4 * HID * 2);
    bf16* cWh     = (bf16*)alloc((size_t)DMP * KP2 * 2);
    bf16* cWm     = (bf16*)alloc((size_t)DMP * KP2 * 2);
    float* b1f    = (float*)alloc(G4 * 4);
    float* b1b    = (float*)alloc(G4 * 4);
    float* b2f_   = (float*)alloc(G4 * 4);
    float* b2b_   = (float*)alloc(G4 * 4);
    float* bhp    = (float*)alloc(DMP * 4);
    float* gx_f   = (float*)alloc((size_t)TOK * G4 * 4);
    float* gx_b   = (float*)alloc((size_t)TOK * G4 * 4);
    float* hs_f   = (float*)alloc((size_t)TOK * HID * 4);
    float* hs_b   = (float*)alloc((size_t)TOK * HID * 4);
    float* Hh     = (float*)alloc((size_t)TOK * DMP * 4);
    float* Hm     = (float*)alloc((size_t)TOK * DMP * 4);

    // ---- weight conversion (fp32 -> bf16, zero-padded) ----
    auto cvt = [&](const float* s, bf16* d, int R, int C, int Rp, int Cp) {
        int tot = Rp * Cp;
        hipLaunchKernelGGL(cvt_pad_bf16, dim3(div_up(tot, 256)), dim3(256), 0, stream,
                           s, d, R, C, Rp, Cp);
    };
    cvt(Wih1f, cWih1f, G4, 125, G4, KP1);
    cvt(Wih1b, cWih1b, G4, 125, G4, KP1);
    cvt(Wih2f, cWih2f, G4, KP2, G4, KP2);
    cvt(Wih2b, cWih2b, G4, KP2, G4, KP2);
    cvt(Whh1f, cWhh1f, G4, HID, G4, HID);
    cvt(Whh1b, cWhh1b, G4, HID, G4, HID);
    cvt(Whh2f, cWhh2f, G4, HID, G4, HID);
    cvt(Whh2b, cWhh2b, G4, HID, G4, HID);
    cvt(Wh, cWh, DMV, KP2, DMP, KP2);
    cvt(Wm, cWm, DMV, KP2, DMP, KP2);

    hipLaunchKernelGGL(vec_add, dim3(8), dim3(256), 0, stream, bih1f, bhh1f, b1f, G4);
    hipLaunchKernelGGL(vec_add, dim3(8), dim3(256), 0, stream, bih1b, bhh1b, b1b, G4);
    hipLaunchKernelGGL(vec_add, dim3(8), dim3(256), 0, stream, bih2f, bhh2f, b2f_, G4);
    hipLaunchKernelGGL(vec_add, dim3(8), dim3(256), 0, stream, bih2b, bhh2b, b2b_, G4);
    hipLaunchKernelGGL(pad_f32, dim3(1), dim3(256), 0, stream, bh, bhp, DMV, DMP);

    // ---- embeddings ----
    hipLaunchKernelGGL(embed_concat, dim3(TOK), dim3(KP1), 0, stream,
                       widx, pidx, wemb, pemb, x0);

    // ---- layer 1: gx = x0 @ Wih^T + bias ; scan ; concat ----
    {
        dim3 g(G4 / 64, TOK / 64), b(128);
        hipLaunchKernelGGL(gemm_bf16_wmma, g, b, 0, stream, x0, cWih1f, b1f, gx_f, TOK, G4, KP1, G4);
        hipLaunchKernelGGL(gemm_bf16_wmma, g, b, 0, stream, x0, cWih1b, b1b, gx_b, TOK, G4, KP1, G4);
    }
    hipLaunchKernelGGL(lstm_scan, dim3(2), dim3(1024), 0, stream,
                       gx_f, gx_b, cWhh1f, cWhh1b, hs_f, hs_b);
    hipLaunchKernelGGL(concat_bf16, dim3(div_up(TOK * HID, 256)), dim3(256), 0, stream,
                       hs_f, hs_b, x1);

    // ---- layer 2 ----
    {
        dim3 g(G4 / 64, TOK / 64), b(128);
        hipLaunchKernelGGL(gemm_bf16_wmma, g, b, 0, stream, x1, cWih2f, b2f_, gx_f, TOK, G4, KP2, G4);
        hipLaunchKernelGGL(gemm_bf16_wmma, g, b, 0, stream, x1, cWih2b, b2b_, gx_b, TOK, G4, KP2, G4);
    }
    hipLaunchKernelGGL(lstm_scan, dim3(2), dim3(1024), 0, stream,
                       gx_f, gx_b, cWhh2f, cWhh2b, hs_f, hs_b);
    hipLaunchKernelGGL(concat_bf16, dim3(div_up(TOK * HID, 256)), dim3(256), 0, stream,
                       hs_f, hs_b, x2);

    // ---- scorer projections: Hh = x2@Wh^T + bh ; Hm = x2@Wm^T ----
    {
        dim3 g(DMP / 64, TOK / 64), b(128);
        hipLaunchKernelGGL(gemm_bf16_wmma, g, b, 0, stream, x2, cWh, bhp, Hh, TOK, DMP, KP2, DMP);
        hipLaunchKernelGGL(gemm_bf16_wmma, g, b, 0, stream, x2, cWm, (const float*)nullptr, Hm, TOK, DMP, KP2, DMP);
    }

    // ---- biaffine relu scores ----
    hipLaunchKernelGGL(biaffine_scores, dim3(TOK / 16, TOK / 16), dim3(16, 16), 0, stream,
                       Hh, Hm, W2, b2, scores);
}
